// GMHAtt_21148418965656
// MI455X (gfx1250) — matrix operand order repaired and checked by reading, not compiled, not run on hardware
//
#include <hip/hip_runtime.h>

#define BB 4
#define SS 1024
#define HH 1024
#define NH 16
#define DD 64

typedef __attribute__((ext_vector_type(16))) __bf16 v16bf;
typedef __attribute__((ext_vector_type(8)))  float  v8f;
typedef unsigned short u16;
typedef unsigned int   u32;

union Frag {
  v16bf bf;
  uint4 q[2];
};

static __device__ __forceinline__ u16 f2bf(float f) {
  u32 u = __float_as_uint(f);
  u32 r = u + 0x7FFFu + ((u >> 16) & 1u);
  return (u16)(r >> 16);
}

// async global->LDS copy of 16 bytes per lane (+byte offset applied to BOTH sides)
static __device__ __forceinline__ void async_b128(u32 lds_addr, const void* gaddr) {
  asm volatile("global_load_async_to_lds_b128 %0, %1, off"
               :: "v"(lds_addr), "v"(gaddr) : "memory");
}
static __device__ __forceinline__ void async_b128_off16(u32 lds_addr, const void* gaddr) {
  asm volatile("global_load_async_to_lds_b128 %0, %1, off offset:16"
               :: "v"(lds_addr), "v"(gaddr) : "memory");
}
static __device__ __forceinline__ void async_b128_off32(u32 lds_addr, const void* gaddr) {
  asm volatile("global_load_async_to_lds_b128 %0, %1, off offset:32"
               :: "v"(lds_addr), "v"(gaddr) : "memory");
}
static __device__ __forceinline__ void async_b128_off48(u32 lds_addr, const void* gaddr) {
  asm volatile("global_load_async_to_lds_b128 %0, %1, off offset:48"
               :: "v"(lds_addr), "v"(gaddr) : "memory");
}
static __device__ __forceinline__ void wait_async0() {
  asm volatile("s_wait_asynccnt 0" ::: "memory");
}
static __device__ __forceinline__ u32 lds_off(const void* p) {
  return (u32)(size_t)p;  // flat LDS pointer: addr[31:0] is the LDS address
}

// ---------------- elementwise f32 -> bf16 ----------------
__global__ void k_cvt_bf16(const float* __restrict__ x, u16* __restrict__ y, int n) {
  int i = blockIdx.x * blockDim.x + threadIdx.x;
  if (i < n) y[i] = f2bf(x[i]);
}

// ---------------- mean over S ----------------
__global__ void k_mean(const float* __restrict__ k, const float* __restrict__ q,
                       float* __restrict__ mk, float* __restrict__ mq) {
  int h = blockIdx.x * blockDim.x + threadIdx.x;
  int b = blockIdx.y;
  const float* x = (blockIdx.z == 0) ? k : q;
  float* m = (blockIdx.z == 0) ? mk : mq;
  float s = 0.f;
  const float* p = x + (size_t)b * SS * HH + h;
  for (int t = 0; t < SS; ++t) s += p[(size_t)t * HH];
  m[b * HH + h] = s * (1.0f / SS);
}

// ---------------- c[b,h] = sum_kk mean[b,kk] * W[kk,h] ----------------
__global__ void k_cproj(const float* __restrict__ mk, const float* __restrict__ mq,
                        const float* __restrict__ Wuk, const float* __restrict__ Wuq,
                        float* __restrict__ ck, float* __restrict__ cq) {
  int h = blockIdx.x * blockDim.x + threadIdx.x;
  int b = blockIdx.y;
  const float* m = (blockIdx.z == 0) ? mk : mq;
  const float* W = (blockIdx.z == 0) ? Wuk : Wuq;
  float* c = (blockIdx.z == 0) ? ck : cq;
  float s = 0.f;
  for (int kk = 0; kk < HH; ++kk) s += m[b * HH + kk] * W[(size_t)kk * HH + h];
  c[b * HH + h] = s;
}

// ---------------- cd[z*B+b] = dot(c[b,:], wt); z=0 -> q ----------------
__global__ void k_cdot(const float* __restrict__ cq, const float* __restrict__ ck,
                       const float* __restrict__ wtcq, const float* __restrict__ wtck,
                       float* __restrict__ cd) {
  __shared__ float red[256];
  int b = blockIdx.x, z = blockIdx.y, t = threadIdx.x;
  const float* c = (z == 0) ? cq : ck;
  const float* w = (z == 0) ? wtcq : wtck;
  float s = 0.f;
  for (int i = t; i < HH; i += 256) s += c[b * HH + i] * w[i];
  red[t] = s; __syncthreads();
  for (int o = 128; o > 0; o >>= 1) { if (t < o) red[t] += red[t + o]; __syncthreads(); }
  if (t == 0) cd[z * BB + b] = red[0];
}

// ---------------- gating: lam = sigmoid(xp.wt + cdot); out = blend (bf16) ----------------
__global__ void k_gate(const float* __restrict__ qp, const float* __restrict__ kp,
                       const float* __restrict__ cq, const float* __restrict__ ck,
                       const float* __restrict__ wtq, const float* __restrict__ wtk,
                       const float* __restrict__ cd,
                       u16* __restrict__ qg, u16* __restrict__ kg) {
  __shared__ float red[256];
  int s = blockIdx.x, b = blockIdx.y, z = blockIdx.z, t = threadIdx.x;
  const float* xp = (z == 0) ? qp : kp;
  const float* c  = (z == 0) ? cq : ck;
  const float* w  = (z == 0) ? wtq : wtk;
  u16* out        = (z == 0) ? qg : kg;
  size_t row = ((size_t)b * SS + s) * HH;
  float acc = 0.f;
  for (int i = t; i < HH; i += 256) acc += xp[row + i] * w[i];
  red[t] = acc; __syncthreads();
  for (int o = 128; o > 0; o >>= 1) { if (t < o) red[t] += red[t + o]; __syncthreads(); }
  float lam = 1.f / (1.f + expf(-(red[0] + cd[z * BB + b])));
  for (int i = t; i < HH; i += 256) {
    float v = (1.f - lam) * xp[row + i] + lam * c[b * HH + i];
    out[row + i] = f2bf(v);
  }
}

// ---------------- WMMA bf16 GEMM: C[M,N] = A[M,K] @ B[K,N] ----------------
// block: 128 thr = 4 waves; block tile 64(M) x 64(N); wave tile 16 x 64.
template <bool OUT_BF16>
__global__ void __launch_bounds__(128)
k_gemm(const u16* __restrict__ A, const u16* __restrict__ Bm,
       void* __restrict__ Cout, int M, int N, int K) {
  __shared__ u16 As[64 * 32];   // [m][k]
  __shared__ u16 Bt[64 * 32];   // [n][k] (transposed)
  const int tid = threadIdx.x;
  const int wave = tid >> 5;
  const int lane = tid & 31;
  const int bm = blockIdx.y * 64;
  const int bn = blockIdx.x * 64;
  v8f acc[4] = {};

  for (int k0 = 0; k0 < K; k0 += 32) {
    __syncthreads();
    { // stage A 64x32 via async global->LDS DMA (32B per thread)
      int r = tid >> 1, half = tid & 1;
      const u16* g = A + (size_t)(bm + r) * K + k0 + half * 16;
      u32 l = lds_off(&As[r * 32 + half * 16]);
      async_b128(l, g);
      async_b128_off16(l, g);
    }
    { // stage B transposed: 32(k) x 64(n) -> Bt[n][k]
      int kk = tid >> 2, seg = tid & 3;
      const uint4* g = reinterpret_cast<const uint4*>(Bm + (size_t)(k0 + kk) * N + bn + seg * 16);
      u16 tmp[16];
      *reinterpret_cast<uint4*>(&tmp[0]) = g[0];
      *reinterpret_cast<uint4*>(&tmp[8]) = g[1];
#pragma unroll
      for (int i = 0; i < 16; ++i) Bt[(seg * 16 + i) * 32 + kk] = tmp[i];
    }
    wait_async0();
    __syncthreads();
    if (k0 + 32 < K) {
      int r = tid >> 1, half = tid & 1;
      __builtin_prefetch(A + (size_t)(bm + r) * K + (k0 + 32) + half * 16, 0, 0);
      int kk = tid >> 2, seg = tid & 3;
      __builtin_prefetch(Bm + (size_t)(k0 + 32 + kk) * N + bn + seg * 16, 0, 0);
    }
    Frag a;
    {
      int m = lane & 15, ko = (lane < 16) ? 0 : 8;
      a.q[0] = *reinterpret_cast<const uint4*>(&As[(wave * 16 + m) * 32 + ko]);
      a.q[1] = *reinterpret_cast<const uint4*>(&As[(wave * 16 + m) * 32 + ko + 16]);
    }
#pragma unroll
    for (int sub = 0; sub < 4; ++sub) {
      Frag bfr;
      int n = sub * 16 + (lane & 15);
      int ko = (lane < 16) ? 0 : 8;
      bfr.q[0] = *reinterpret_cast<const uint4*>(&Bt[n * 32 + ko]);
      bfr.q[1] = *reinterpret_cast<const uint4*>(&Bt[n * 32 + ko + 16]);
      acc[sub] = __builtin_amdgcn_wmma_f32_16x16x32_bf16(
          false, a.bf, false, bfr.bf, (short)0, acc[sub], false, false);
    }
  }

  int rb = bm + wave * 16 + ((lane < 16) ? 0 : 8);
  int cb = bn + (lane & 15);
  if (OUT_BF16) {
    u16* C = (u16*)Cout;
#pragma unroll
    for (int sub = 0; sub < 4; ++sub)
#pragma unroll
      for (int r = 0; r < 8; ++r)
        C[(size_t)(rb + r) * N + cb + sub * 16] = f2bf(acc[sub][r]);
  } else {
    float* C = (float*)Cout;
#pragma unroll
    for (int sub = 0; sub < 4; ++sub)
#pragma unroll
      for (int r = 0; r < 8; ++r)
        C[(size_t)(rb + r) * N + cb + sub * 16] = acc[sub][r];
  }
}

// ---------------- flash attention: one (b,h), 64 query rows per block ----------------
__global__ void __launch_bounds__(128)
k_attn(const u16* __restrict__ qg, const u16* __restrict__ kg,
       const u16* __restrict__ vp, const unsigned char* __restrict__ mask,
       u16* __restrict__ out) {
  __shared__ u16 Ks[64 * 64];       // [key][d]
  __shared__ u16 Vt[64 * 64];       // [d][key]
  __shared__ u16 Ps[4][16 * 64];    // per-wave P tile, [m][key]
  const int tid = threadIdx.x, wave = tid >> 5, lane = tid & 31;
  const int qb = blockIdx.x * 64;
  const int h = blockIdx.y;
  const int b = blockIdx.z;
  const size_t base = (size_t)b * SS * HH + (size_t)h * DD;

  Frag qa[2];
  {
    int m = lane & 15;
    size_t rowoff = base + (size_t)(qb + wave * 16 + m) * HH;
#pragma unroll
    for (int f = 0; f < 2; ++f) {
      int ko = f * 32 + ((lane < 16) ? 0 : 8);
      qa[f].q[0] = *reinterpret_cast<const uint4*>(qg + rowoff + ko);
      qa[f].q[1] = *reinterpret_cast<const uint4*>(qg + rowoff + ko + 16);
    }
  }
  v8f o[4] = {};
  float mrow[8], lrow[8];
#pragma unroll
  for (int r = 0; r < 8; ++r) { mrow[r] = -1e30f; lrow[r] = 0.f; }
  const float scale = 0.125f;  // 1/sqrt(64)

  for (int kt = 0; kt < SS / 64; ++kt) {
    __syncthreads();
    { // stage K tile [key][d] via async global->LDS DMA (64B per thread)
      int key = tid >> 1, half = tid & 1;
      const u16* g = kg + base + (size_t)(kt * 64 + key) * HH + half * 32;
      u32 l = lds_off(&Ks[key * 64 + half * 32]);
      async_b128(l, g);
      async_b128_off16(l, g);
      async_b128_off32(l, g);
      async_b128_off48(l, g);
    }
    { // stage V transposed [d][key] (DMA cannot transpose; VGPR round-trip)
      int key = tid >> 1, half = tid & 1;
      size_t roff = base + (size_t)(kt * 64 + key) * HH + half * 32;
      u16 tmp[32];
#pragma unroll
      for (int i = 0; i < 4; ++i)
        *reinterpret_cast<uint4*>(&tmp[i * 8]) =
            *reinterpret_cast<const uint4*>(vp + roff + i * 8);
#pragma unroll
      for (int i = 0; i < 32; ++i) Vt[(half * 32 + i) * 64 + key] = tmp[i];
    }
    wait_async0();
    __syncthreads();

    // scores S = Q @ K^T * scale  (with mask)
    v8f sc[4];
#pragma unroll
    for (int sub = 0; sub < 4; ++sub) {
      v8f s = {};
#pragma unroll
      for (int c = 0; c < 2; ++c) {
        Frag bfr;
        int n = sub * 16 + (lane & 15);
        int ko = c * 32 + ((lane < 16) ? 0 : 8);
        bfr.q[0] = *reinterpret_cast<const uint4*>(&Ks[n * 64 + ko]);
        bfr.q[1] = *reinterpret_cast<const uint4*>(&Ks[n * 64 + ko + 16]);
        s = __builtin_amdgcn_wmma_f32_16x16x32_bf16(
            false, qa[c].bf, false, bfr.bf, (short)0, s, false, false);
      }
      sc[sub] = s;
      int t = kt * 64 + sub * 16 + (lane & 15);
      bool mk = mask[(size_t)b * SS + t] != 0;
#pragma unroll
      for (int r = 0; r < 8; ++r) {
        float v = sc[sub][r] * scale;
        sc[sub][r] = mk ? -1e9f : v;
      }
    }

    // online softmax update
#pragma unroll
    for (int r = 0; r < 8; ++r) {
      float v = fmaxf(fmaxf(sc[0][r], sc[1][r]), fmaxf(sc[2][r], sc[3][r]));
      v = fmaxf(v, __shfl_xor(v, 1, 32));
      v = fmaxf(v, __shfl_xor(v, 2, 32));
      v = fmaxf(v, __shfl_xor(v, 4, 32));
      v = fmaxf(v, __shfl_xor(v, 8, 32));
      float nm = fmaxf(mrow[r], v);
      float alpha = expf(mrow[r] - nm);
      float rs = 0.f;
#pragma unroll
      for (int sub = 0; sub < 4; ++sub) {
        float p = expf(sc[sub][r] - nm);
        sc[sub][r] = p;
        rs += p;
      }
      rs += __shfl_xor(rs, 1, 32);
      rs += __shfl_xor(rs, 2, 32);
      rs += __shfl_xor(rs, 4, 32);
      rs += __shfl_xor(rs, 8, 32);
      lrow[r] = lrow[r] * alpha + rs;
      mrow[r] = nm;
#pragma unroll
      for (int sub = 0; sub < 4; ++sub) o[sub][r] *= alpha;
    }

    // P (C-layout) -> per-wave LDS in A-layout-friendly row-major [m][key]
    {
      u16* P = Ps[wave];
      int m0 = (lane < 16) ? 0 : 8;
      int col = lane & 15;
#pragma unroll
      for (int sub = 0; sub < 4; ++sub)
#pragma unroll
        for (int r = 0; r < 8; ++r)
          P[(m0 + r) * 64 + sub * 16 + col] = f2bf(sc[sub][r]);
    }
    Frag pa[2];
    {
      const u16* P = Ps[wave];
      int m = lane & 15;
#pragma unroll
      for (int f = 0; f < 2; ++f) {
        int ko = f * 32 + ((lane < 16) ? 0 : 8);
        pa[f].q[0] = *reinterpret_cast<const uint4*>(&P[m * 64 + ko]);
        pa[f].q[1] = *reinterpret_cast<const uint4*>(&P[m * 64 + ko + 16]);
      }
    }
    // O += P @ V
#pragma unroll
    for (int sub = 0; sub < 4; ++sub) {
#pragma unroll
      for (int c = 0; c < 2; ++c) {
        Frag bfr;
        int n = sub * 16 + (lane & 15);           // d column
        int ko = c * 32 + ((lane < 16) ? 0 : 8);  // key
        bfr.q[0] = *reinterpret_cast<const uint4*>(&Vt[n * 64 + ko]);
        bfr.q[1] = *reinterpret_cast<const uint4*>(&Vt[n * 64 + ko + 16]);
        o[sub] = __builtin_amdgcn_wmma_f32_16x16x32_bf16(
            false, pa[c].bf, false, bfr.bf, (short)0, o[sub], false, false);
      }
    }
  }

  // finalize: O / l -> bf16 into [B,S,H] layout (head h occupies cols h*64..)
  {
    int rb = qb + wave * 16 + ((lane < 16) ? 0 : 8);
    int col = lane & 15;
#pragma unroll
    for (int sub = 0; sub < 4; ++sub)
#pragma unroll
      for (int r = 0; r < 8; ++r) {
        float v = o[sub][r] / lrow[r];
        out[base + (size_t)(rb + r) * HH + sub * 16 + col] = f2bf(v);
      }
  }
}

extern "C" void kernel_launch(void* const* d_in, const int* in_sizes, int n_in,
                              void* d_out, int out_size, void* d_ws, size_t ws_size,
                              hipStream_t stream) {
  const float* v = (const float*)d_in[0];
  const float* k = (const float*)d_in[1];
  const float* q = (const float*)d_in[2];
  const unsigned char* mask = (const unsigned char*)d_in[3];
  const float* Wv = (const float*)d_in[4];
  const float* Wk = (const float*)d_in[5];
  const float* Wq = (const float*)d_in[6];
  const float* Wm = (const float*)d_in[7];
  const float* Wuk = (const float*)d_in[8];
  const float* Wuq = (const float*)d_in[9];
  const float* wtq = (const float*)d_in[10];
  const float* wtk = (const float*)d_in[11];
  const float* wtcq = (const float*)d_in[12];
  const float* wtck = (const float*)d_in[13];
  float* outp = (float*)d_out;

  char* wsb = (char*)d_ws;
  size_t off = 0;
  auto alloc = [&](size_t bytes) -> char* {
    char* p = wsb + off;
    off += (bytes + 255) & ~size_t(255);
    return p;
  };
  const size_t NTOK = (size_t)BB * SS;   // 4096
  const size_t NEL = NTOK * HH;          // 4M

  u16* vB = (u16*)alloc(NEL * 2);
  u16* kB = (u16*)alloc(NEL * 2);
  u16* qB = (u16*)alloc(NEL * 2);
  u16* WvB = (u16*)alloc((size_t)HH * HH * 2);
  u16* WkB = (u16*)alloc((size_t)HH * HH * 2);
  u16* WqB = (u16*)alloc((size_t)HH * HH * 2);
  u16* WmB = (u16*)alloc((size_t)HH * HH * 2);
  float* qp = (float*)alloc(NEL * 4);
  float* kp = (float*)alloc(NEL * 4);
  u16* vpB = (u16*)alloc(NEL * 2);
  u16* qgB = (u16*)alloc(NEL * 2);
  u16* kgB = (u16*)alloc(NEL * 2);
  u16* attnB = (u16*)alloc(NEL * 2);
  float* mK = (float*)alloc((size_t)BB * HH * 4);
  float* mQ = (float*)alloc((size_t)BB * HH * 4);
  float* cK = (float*)alloc((size_t)BB * HH * 4);
  float* cQ = (float*)alloc((size_t)BB * HH * 4);
  float* cd = (float*)alloc(2 * BB * 4);

  // bf16 conversions
  k_cvt_bf16<<<(int)((NEL + 255) / 256), 256, 0, stream>>>(v, vB, (int)NEL);
  k_cvt_bf16<<<(int)((NEL + 255) / 256), 256, 0, stream>>>(k, kB, (int)NEL);
  k_cvt_bf16<<<(int)((NEL + 255) / 256), 256, 0, stream>>>(q, qB, (int)NEL);
  int nw = HH * HH;
  k_cvt_bf16<<<(nw + 255) / 256, 256, 0, stream>>>(Wv, WvB, nw);
  k_cvt_bf16<<<(nw + 255) / 256, 256, 0, stream>>>(Wk, WkB, nw);
  k_cvt_bf16<<<(nw + 255) / 256, 256, 0, stream>>>(Wq, WqB, nw);
  k_cvt_bf16<<<(nw + 255) / 256, 256, 0, stream>>>(Wm, WmB, nw);

  // context path
  k_mean<<<dim3(HH / 256, BB, 2), 256, 0, stream>>>(k, q, mK, mQ);
  k_cproj<<<dim3(HH / 256, BB, 2), 256, 0, stream>>>(mK, mQ, Wuk, Wuq, cK, cQ);
  k_cdot<<<dim3(BB, 2), 256, 0, stream>>>(cQ, cK, wtcq, wtck, cd);

  // projections (WMMA)
  dim3 ggrid(HH / 64, (unsigned)(NTOK / 64));
  k_gemm<false><<<ggrid, 128, 0, stream>>>(qB, WqB, qp, (int)NTOK, HH, HH);
  k_gemm<false><<<ggrid, 128, 0, stream>>>(kB, WkB, kp, (int)NTOK, HH, HH);
  k_gemm<true><<<ggrid, 128, 0, stream>>>(vB, WvB, vpB, (int)NTOK, HH, HH);

  // gating
  k_gate<<<dim3(SS, BB, 2), 256, 0, stream>>>(qp, kp, cQ, cK, wtq, wtk, cd, qgB, kgB);

  // attention (WMMA flash)
  k_attn<<<dim3(SS / 64, NH, BB), 128, 0, stream>>>(qgB, kgB, vpB, mask, attnB);

  // output projection (WMMA) -> f32 d_out
  k_gemm<false><<<ggrid, 128, 0, stream>>>(attnB, WmB, outp, (int)NTOK, HH, HH);
}
